// VectorQuantize2d_52312701665799
// MI455X (gfx1250) — compile-verified, matrix operand order checked by reading, not run on hardware
//
#include <hip/hip_runtime.h>
#include <hip/hip_bf16.h>

// Dimensions from reference
#define BB   16
#define CC   512
#define HWp  4096          // H*W = 64*64
#define DD   64
#define KCB  2048
#define NTOK (BB * HWp)    // 65536

typedef float    v2f  __attribute__((ext_vector_type(2)));
typedef float    v8f  __attribute__((ext_vector_type(8)));
typedef _Float16 v8h  __attribute__((ext_vector_type(8)));
typedef _Float16 v16h __attribute__((ext_vector_type(16)));

__device__ __forceinline__ v8f wmma_f32(v2f a, v2f b, v8f c) {
  // D = A(16x4) * B(4x16) + C, fp32 WMMA
  return __builtin_amdgcn_wmma_f32_16x16x4_f32(
      false, a, false, b, (short)0, c, false, false);
}

__device__ __forceinline__ v8f wmma_f16(v16h a, v16h b, v8f c) {
  // D = A(16x32) * B(32x16) + C, f16 inputs, fp32 accumulate
  return __builtin_amdgcn_wmma_f32_16x16x32_f16(
      false, a, false, b, (short)0, c, false, false);
}

__device__ __forceinline__ v8f zero8() {
  v8f z;
  #pragma unroll
  for (int i = 0; i < 8; ++i) z[i] = 0.0f;
  return z;
}

// Build a v16h fragment from a contiguous 64-element row:
// lanes 0-15 take K = [o..o+7] and [o+16..o+23]; lanes 16-31 the +8 halves.
// Caller passes o = 32*ks + 8*half.
__device__ __forceinline__ v16h frag16(const _Float16* row, int o) {
  v8h lo = *(const v8h*)(row + o);
  v8h hi = *(const v8h*)(row + o + 16);
  return __builtin_shufflevector(lo, hi, 0, 1, 2, 3, 4, 5, 6, 7,
                                         8, 9, 10, 11, 12, 13, 14, 15);
}

// ---------------- weight norm: w[row,:] = g[row] * v[row,:] / ||v[row,:]|| ----
__global__ void __launch_bounds__(256)
wn_kernel(const float* __restrict__ v, const float* __restrict__ g,
          float* __restrict__ w, int cols) {
  const int row = blockIdx.x;
  const int tid = threadIdx.x;
  __shared__ float red[256];
  float ss = 0.0f;
  for (int i = tid; i < cols; i += 256) {
    float x = v[row * cols + i];
    ss += x * x;
  }
  red[tid] = ss;
  __syncthreads();
  for (int s = 128; s > 0; s >>= 1) {
    if (tid < s) red[tid] += red[tid + s];
    __syncthreads();
  }
  const float scale = g[row] / sqrtf(red[0]);
  for (int i = tid; i < cols; i += 256)
    w[row * cols + i] = v[row * cols + i] * scale;
}

// -------- codebook squared norms + f16 copy ---------------------------------
__global__ void __launch_bounds__(256)
cbnorm_kernel(const float* __restrict__ cb, float* __restrict__ cbn,
              _Float16* __restrict__ cb_h) {
  const int r = blockIdx.x * 256 + threadIdx.x;   // 2048 rows
  const float* row = cb + (size_t)r * DD;
  _Float16* hrow = cb_h + (size_t)r * DD;
  float ss = 0.0f;
  #pragma unroll 8
  for (int i = 0; i < DD; ++i) {
    float x = row[i];
    ss += x * x;
    hrow[i] = (_Float16)x;
  }
  cbn[r] = ss;
}

// ---------------- in_proj: z_e[b][d][hw] = sum_c Wi[d][c] * z[b][c][hw] + in_b
// Per block: M=64 (d) x N=32 (hw); 8 waves, each one 16x16 tile; K=512 step 4.
// Writes fp32 z_e d-major and an f16 token-major copy for the distance GEMM.
__global__ void __launch_bounds__(256)
in_proj_kernel(const float* __restrict__ z, const float* __restrict__ Wi,
               const float* __restrict__ in_b, float* __restrict__ z_e,
               _Float16* __restrict__ z_e16) {
  const int lane = threadIdx.x & 31;
  const int w    = threadIdx.x >> 5;
  const int half = lane >> 4;
  const int l15  = lane & 15;
  const int b    = blockIdx.y;
  const int hw0  = blockIdx.x * 32 + (w >> 2) * 16;
  const int m0   = (w & 3) * 16;

  const float* zb = z + (size_t)b * CC * HWp;
  v8f acc = zero8();

  const int m = m0 + l15;
  const int n = hw0 + l15;
  #pragma unroll 4
  for (int k = 0; k < CC; k += 4) {
    const int kk = k + half * 2;
    v2f a = *(const v2f*)(Wi + (size_t)m * CC + kk);  // A[m][kk], A[m][kk+1]
    v2f bf;
    bf.x = zb[(size_t)kk * HWp + n];                  // B[kk][n]
    bf.y = zb[(size_t)(kk + 1) * HWp + n];            // B[kk+1][n]
    acc = wmma_f32(a, bf, acc);
  }
  const size_t token = (size_t)b * HWp + n;
  #pragma unroll
  for (int r = 0; r < 8; ++r) {
    const int mm = m0 + r + half * 8;
    const float val = acc[r] + in_b[mm];
    z_e[((size_t)b * DD + mm) * HWp + n] = val;     // d-major fp32
    z_e16[token * DD + mm] = (_Float16)val;         // token-major f16
  }
}

// ---------------- distance + argmin (f16 WMMA + exact fp32 top-2 refine) ----
// Wave handles 16 tokens; loops over 128 code tiles of 16; K=64 as 2x(16x16x32).
// argmin_n { ||e_n||^2 - 2 z_e . e_n }  (token norm is a constant, dropped)
__global__ void __launch_bounds__(256)
dist_argmin_kernel(const float* __restrict__ z_e32,
                   const _Float16* __restrict__ z_e16,
                   const _Float16* __restrict__ cb_h,
                   const float* __restrict__ cb,
                   const float* __restrict__ cbn, int* __restrict__ indices) {
  const int lane = threadIdx.x & 31;
  const int w    = threadIdx.x >> 5;
  const int half = lane >> 4;
  const int l15  = lane & 15;
  const int t0   = (blockIdx.x * 8 + w) * 16;   // 16-token tile (never crosses b)
  const int b    = t0 >> 12;
  const int hw0  = t0 & (HWp - 1);

  // A fragments for the 16 tokens, cached across all code tiles (16 VGPRs).
  const _Float16* arow = z_e16 + (size_t)(t0 + l15) * DD;
  v16h afrag[2];
  #pragma unroll
  for (int ks = 0; ks < 2; ++ks) afrag[ks] = frag16(arow, ks * 32 + half * 8);

  float best_val[8];
  int   best_idx[8];
  #pragma unroll
  for (int r = 0; r < 8; ++r) { best_val[r] = 3.4e38f; best_idx[r] = 0; }

  for (int n0 = 0; n0 < KCB; n0 += 16) {
    const int n = n0 + l15;
    const _Float16* brow = cb_h + (size_t)n * DD;   // B[k][n] = cb[n][k]
    v8f acc = zero8();
    acc = wmma_f16(afrag[0], frag16(brow, 0 + half * 8), acc);
    acc = wmma_f16(afrag[1], frag16(brow, 32 + half * 8), acc);
    const float cn = cbn[n];
    #pragma unroll
    for (int r = 0; r < 8; ++r) {
      const float val = cn - 2.0f * acc[r];
      if (val < best_val[r]) { best_val[r] = val; best_idx[r] = n; }
    }
  }

  // Cross-lane: candidates for token m live in 16 lanes of one half-wave.
  __shared__ float s_val[8][16][16];
  __shared__ int   s_idx[8][16][16];
  #pragma unroll
  for (int r = 0; r < 8; ++r) {
    const int ml = r + half * 8;     // token-in-tile 0..15
    s_val[w][ml][l15] = best_val[r];
    s_idx[w][ml][l15] = best_idx[r];
  }
  __syncthreads();
  if (lane < 16) {
    // top-2 candidates across the 16 residue classes
    float bv0 = 3.4e38f, bv1 = 3.4e38f;
    int   bi0 = 0,       bi1 = 0;
    #pragma unroll
    for (int j = 0; j < 16; ++j) {
      const float vj = s_val[w][lane][j];
      const int   ij = s_idx[w][lane][j];
      if (vj < bv0)      { bv1 = bv0; bi1 = bi0; bv0 = vj; bi0 = ij; }
      else if (vj < bv1) { bv1 = vj; bi1 = ij; }
    }
    // exact fp32 rescore of the two candidates
    const int hw = hw0 + lane;
    const float* zeb = z_e32 + (size_t)b * DD * HWp + hw;
    const float* c0 = cb + (size_t)bi0 * DD;
    const float* c1 = cb + (size_t)bi1 * DD;
    float d0 = 0.0f, d1 = 0.0f;
    #pragma unroll 8
    for (int d = 0; d < DD; ++d) {
      const float ze = zeb[(size_t)d * HWp];
      const float a0 = ze - c0[d];
      const float a1 = ze - c1[d];
      d0 += a0 * a0;
      d1 += a1 * a1;
    }
    int best = bi0;
    if (d1 < d0 || (d1 == d0 && bi1 < bi0)) best = bi1;
    indices[t0 + lane] = best;
  }
}

// ---------------- gather z_q + loss partials --------------------------------
__global__ void __launch_bounds__(256)
gather_loss_kernel(const float* __restrict__ z_e, const float* __restrict__ cb,
                   const int* __restrict__ indices, float* __restrict__ z_q,
                   float* __restrict__ partial) {
  const int tid = threadIdx.x;
  const size_t e = (size_t)blockIdx.x * 256 + tid;   // e = ((b*64+d)<<12) + hw
  const int hw = (int)(e & (HWp - 1));
  const int d  = (int)((e >> 12) & (DD - 1));
  const int b  = (int)(e >> 18);
  const int token = b * HWp + hw;
  const int idx = indices[token];
  const float zq = cb[(size_t)idx * DD + d];
  const float ze = z_e[e];
  z_q[e] = zq;
  const float diff = ze - zq;

  __shared__ float red[256];
  red[tid] = diff * diff;
  __syncthreads();
  for (int s = 128; s > 0; s >>= 1) {
    if (tid < s) red[tid] += red[tid + s];
    __syncthreads();
  }
  if (tid == 0) partial[blockIdx.x] = red[0];
}

// ---------------- out_proj: out[b][c][hw] = sum_d Wo[c][d]*z_q[b][d][hw]+out_b
__global__ void __launch_bounds__(256)
out_proj_kernel(const float* __restrict__ z_q, const float* __restrict__ Wo,
                const float* __restrict__ out_b, float* __restrict__ out) {
  const int lane = threadIdx.x & 31;
  const int w    = threadIdx.x >> 5;
  const int half = lane >> 4;
  const int l15  = lane & 15;
  const int b    = blockIdx.z;
  const int m0   = blockIdx.y * 64 + (w & 3) * 16;
  const int hw0  = blockIdx.x * 32 + (w >> 2) * 16;

  const float* zqb = z_q + (size_t)b * DD * HWp;
  v8f acc = zero8();

  const int m = m0 + l15;
  const int n = hw0 + l15;
  #pragma unroll
  for (int k = 0; k < DD; k += 4) {
    const int kk = k + half * 2;
    v2f a = *(const v2f*)(Wo + (size_t)m * DD + kk);
    v2f bf;
    bf.x = zqb[(size_t)kk * HWp + n];
    bf.y = zqb[(size_t)(kk + 1) * HWp + n];
    acc = wmma_f32(a, bf, acc);
  }
  #pragma unroll
  for (int r = 0; r < 8; ++r) {
    const int mm = m0 + r + half * 8;
    out[((size_t)b * CC + mm) * HWp + n] = acc[r] + out_b[mm];
  }
}

// ---------------- final loss reduction --------------------------------------
__global__ void __launch_bounds__(256)
loss_final_kernel(const float* __restrict__ partial, int nPartial,
                  float* __restrict__ loss) {
  __shared__ float red[256];
  const int tid = threadIdx.x;
  float s = 0.0f;
  for (int i = tid; i < nPartial; i += 256) s += partial[i];
  red[tid] = s;
  __syncthreads();
  for (int st = 128; st > 0; st >>= 1) {
    if (tid < st) red[tid] += red[tid + st];
    __syncthreads();
  }
  // vq_loss = codebook_loss + 0.25*commitment = 1.25 * mean((z_e - z_q)^2)
  if (tid == 0) loss[0] = 1.25f * red[0] / (float)((size_t)NTOK * DD);
}

extern "C" void kernel_launch(void* const* d_in, const int* in_sizes, int n_in,
                              void* d_out, int out_size, void* d_ws, size_t ws_size,
                              hipStream_t stream) {
  const float* z        = (const float*)d_in[0];  // [16,512,64,64]
  const float* codebook = (const float*)d_in[1];  // [2048,64]
  const float* in_v     = (const float*)d_in[2];  // [64,512]
  const float* in_g     = (const float*)d_in[3];  // [64]
  const float* in_b     = (const float*)d_in[4];  // [64]
  const float* out_v    = (const float*)d_in[5];  // [512,64]
  const float* out_g    = (const float*)d_in[6];  // [512]
  const float* out_b    = (const float*)d_in[7];  // [512]

  float* out     = (float*)d_out;                         // [16,512,64,64]
  int*   indices = (int*)(out + (size_t)BB * CC * HWp);   // [16,64,64] int32
  float* loss    = (float*)(indices + NTOK);              // scalar

  float* ws      = (float*)d_ws;
  float* Wi      = ws;                         // 64*512
  float* Wo      = Wi + (size_t)DD * CC;       // 512*64
  float* cbn     = Wo + (size_t)CC * DD;       // 2048
  float* partial = cbn + KCB;                  // 16384
  float* z_e     = partial + 16384;            // 16*64*4096 fp32, d-major
  float* z_q     = z_e + (size_t)BB * DD * HWp;
  _Float16* cb_h  = (_Float16*)(z_q + (size_t)BB * DD * HWp);   // 2048*64 f16
  _Float16* z_e16 = cb_h + (size_t)KCB * DD;                    // 65536*64 f16

  wn_kernel<<<dim3(DD),  dim3(256), 0, stream>>>(in_v,  in_g,  Wi, CC);
  wn_kernel<<<dim3(CC),  dim3(256), 0, stream>>>(out_v, out_g, Wo, DD);
  cbnorm_kernel<<<dim3(KCB / 256), dim3(256), 0, stream>>>(codebook, cbn, cb_h);

  in_proj_kernel<<<dim3(HWp / 32, BB), dim3(256), 0, stream>>>(z, Wi, in_b,
                                                               z_e, z_e16);
  dist_argmin_kernel<<<dim3(NTOK / 128), dim3(256), 0, stream>>>(
      z_e, z_e16, cb_h, codebook, cbn, indices);
  gather_loss_kernel<<<dim3((BB * DD * HWp) / 256), dim3(256), 0, stream>>>(
      z_e, codebook, indices, z_q, partial);
  out_proj_kernel<<<dim3(HWp / 32, CC / 64, BB), dim3(256), 0, stream>>>(
      z_q, Wo, out_b, out);
  loss_final_kernel<<<1, 256, 0, stream>>>(partial, (BB * DD * HWp) / 256, loss);
}
